// TransformerDecoder_55731495633313
// MI455X (gfx1250) — compile-verified, matrix operand order checked by reading, not run on hardware
//
#include <hip/hip_runtime.h>
#include <math.h>

typedef __attribute__((ext_vector_type(16))) _Float16 v16h;
typedef __attribute__((ext_vector_type(8)))  _Float16 v8h;
typedef __attribute__((ext_vector_type(8)))  float    v8f;

#define LL 12
#define DD 512
#define HH 8
#define DHH 64
#define FF 2048
#define BB 4
#define SS 1024
#define MM (BB * SS)   // 4096 rows

#define MODE_F32      0
#define MODE_F16      1
#define MODE_F16_RELU 2
#define MODE_VT       3   // f16, V transposed per head: [B,H,DH,S]

// ---------------------------------------------------------------------------
// GEMM: C[M,N] = A[M,K] @ Wt^T + bias.  A: f16 row-major.  Wt: f16 [N,K].
// Block = 256 thr (8 waves as 4(M) x 2(N)).  Block tile 128x128, K-step 32.
// A tile staged to LDS via global_load_async_to_lds_b128 (ASYNCcnt path);
// B fragments loaded straight from global (weights L2-resident).
// Each wave: 32x64 = 8 x v_wmma_f32_16x16x32_f16 per K-step.
// ---------------------------------------------------------------------------
__global__ void __launch_bounds__(256)
gemm_wmma(const _Float16* __restrict__ A, const _Float16* __restrict__ Bt,
          const float* __restrict__ bias, void* __restrict__ C,
          int M, int N, int K, int mode)
{
    __shared__ alignas(16) _Float16 As[128][40];  // 80B row stride (16B-aligned)

    const int tid  = threadIdx.x;
    const int wave = tid >> 5;
    const int lane = tid & 31;
    const int wm   = wave & 3;        // 0..3 -> 32-row strip
    const int wn   = wave >> 2;       // 0..1 -> 64-col strip
    const int n0   = blockIdx.x * 128;
    const int m0   = blockIdx.y * 128;
    const int half = lane >> 4;
    const int l15  = lane & 15;

    const unsigned asBase = (unsigned)(size_t)&As[0][0];

    // A staging: 128 rows x 32 halves = 512 x 16B chunks; 2 chunks/thread
    const int c0row = tid >> 2,          c0q = tid & 3;
    const int c1row = (tid + 256) >> 2,  c1q = tid & 3;   // (tid+256)&3 == tid&3

    v8f acc[2][4] = {};

    for (int k0 = 0; k0 < K; k0 += 32) {
        {
            const unsigned l0 = asBase + (unsigned)(c0row * 80 + c0q * 16);
            const _Float16* g0 = A + (size_t)(m0 + c0row) * K + k0 + c0q * 8;
            asm volatile("global_load_async_to_lds_b128 %0, %1, off"
                         :: "v"(l0), "v"(g0) : "memory");
            const unsigned l1 = asBase + (unsigned)(c1row * 80 + c1q * 16);
            const _Float16* g1 = A + (size_t)(m0 + c1row) * K + k0 + c1q * 8;
            asm volatile("global_load_async_to_lds_b128 %0, %1, off"
                         :: "v"(l1), "v"(g1) : "memory");
        }
        asm volatile("s_wait_asynccnt 0x0" ::: "memory");
        __syncthreads();

        // A fragments (2 sub-tiles of 16 rows)
        v16h af[2];
        #pragma unroll
        for (int s = 0; s < 2; ++s) {
            const int row = wm * 32 + s * 16 + l15;
            v8h alo = *(const v8h*)&As[row][half * 8];
            v8h ahi = *(const v8h*)&As[row][16 + half * 8];
            #pragma unroll
            for (int j = 0; j < 8; ++j) { af[s][j] = alo[j]; af[s][8 + j] = ahi[j]; }
        }

        // B fragments direct from global Wt[n][k]
        #pragma unroll
        for (int nt = 0; nt < 4; ++nt) {
            const _Float16* bp =
                Bt + (size_t)(n0 + wn * 64 + nt * 16 + l15) * K + k0;
            v8h blo = *(const v8h*)(bp + half * 8);
            v8h bhi = *(const v8h*)(bp + 16 + half * 8);
            v16h bf;
            #pragma unroll
            for (int j = 0; j < 8; ++j) { bf[j] = blo[j]; bf[8 + j] = bhi[j]; }
            acc[0][nt] = __builtin_amdgcn_wmma_f32_16x16x32_f16(
                false, af[0], false, bf, (short)0, acc[0][nt], false, false);
            acc[1][nt] = __builtin_amdgcn_wmma_f32_16x16x32_f16(
                false, af[1], false, bf, (short)0, acc[1][nt], false, false);
        }
        __syncthreads();
    }

    // epilogue
    float*    Cf = (float*)C;
    _Float16* Ch = (_Float16*)C;
    #pragma unroll
    for (int nt = 0; nt < 4; ++nt) {
        const int col = n0 + wn * 64 + nt * 16 + l15;
        const float bv = bias ? bias[col] : 0.0f;
        #pragma unroll
        for (int s = 0; s < 2; ++s) {
            #pragma unroll
            for (int i = 0; i < 8; ++i) {
                const int row = m0 + wm * 32 + s * 16 + half * 8 + i;
                float v = acc[s][nt][i] + bv;
                if (mode == MODE_F32) {
                    Cf[(size_t)row * N + col] = v;
                } else if (mode == MODE_VT) {
                    const int bb = row >> 10, ss = row & (SS - 1);
                    const int hh = col >> 6,  dh = col & 63;
                    Ch[(size_t)((bb * HH + hh) * DHH + dh) * SS + ss] = (_Float16)v;
                } else {
                    if (mode == MODE_F16_RELU) v = v > 0.0f ? v : 0.0f;
                    Ch[(size_t)row * N + col] = (_Float16)v;
                }
            }
        }
    }
}

// ---------------------------------------------------------------------------
// Flash-style attention, all-f16 operands.  Q,K: f16 [B*S, D] (head h at cols
// h*64..).  Vt: f16 [B,H,DH,S].  O: f16 [B*S, D].  One wave per 16 q-rows.
// ---------------------------------------------------------------------------
__global__ void __launch_bounds__(128)
attn_wmma(const _Float16* __restrict__ Q, const _Float16* __restrict__ Kb,
          const _Float16* __restrict__ Vt, _Float16* __restrict__ O,
          int Sq, int Sk, int causal)
{
    __shared__ alignas(16) _Float16 Ps[4][16 * 32];

    const int tid  = threadIdx.x;
    const int wave = tid >> 5;
    const int lane = tid & 31;
    const int nqb  = Sq >> 6;
    const int qb   = blockIdx.x % nqb;
    const int h    = (blockIdx.x / nqb) % HH;
    const int b    = blockIdx.x / (nqb * HH);
    const int qrow0   = qb * 64 + wave * 16;
    const int half    = lane >> 4;
    const int l15     = lane & 15;
    const int colbase = h * DHH;

    // Q fragments (f16 direct loads; softmax scale applied to fp32 scores)
    v16h qf[2];
    {
        const _Float16* qrow = Q + (size_t)(b * Sq + qrow0 + l15) * DD + colbase;
        #pragma unroll
        for (int f = 0; f < 2; ++f) {
            v8h qlo = *(const v8h*)(qrow + f * 32 + half * 8);
            v8h qhi = *(const v8h*)(qrow + f * 32 + 16 + half * 8);
            #pragma unroll
            for (int j = 0; j < 8; ++j) { qf[f][j] = qlo[j]; qf[f][8 + j] = qhi[j]; }
        }
    }

    float mrow[8], lrow[8];
    #pragma unroll
    for (int i = 0; i < 8; ++i) { mrow[i] = -1e30f; lrow[i] = 0.0f; }
    v8f oacc[4] = {};

    const int kbmax = causal ? ((qrow0 + 15) >> 5) : ((Sk >> 5) - 1);
    for (int kb = 0; kb <= kbmax; ++kb) {
        const int ks = kb * 32;

        // ---- S = Q @ K^T (scaled in fp32 after) ----
        v8f sc[2];
        #pragma unroll
        for (int nt = 0; nt < 2; ++nt) {
            const _Float16* krow =
                Kb + (size_t)(b * Sk + ks + nt * 16 + l15) * DD + colbase;
            v8f s = {};
            #pragma unroll
            for (int f = 0; f < 2; ++f) {
                v8h klo = *(const v8h*)(krow + f * 32 + half * 8);
                v8h khi = *(const v8h*)(krow + f * 32 + 16 + half * 8);
                v16h kf;
                #pragma unroll
                for (int j = 0; j < 8; ++j) { kf[j] = klo[j]; kf[8 + j] = khi[j]; }
                s = __builtin_amdgcn_wmma_f32_16x16x32_f16(
                    false, qf[f], false, kf, (short)0, s, false, false);
            }
            sc[nt] = s;
        }
        #pragma unroll
        for (int nt = 0; nt < 2; ++nt)
            #pragma unroll
            for (int i = 0; i < 8; ++i) sc[nt][i] *= 0.125f;  // 1/sqrt(64)

        if (causal) {
            #pragma unroll
            for (int nt = 0; nt < 2; ++nt) {
                const int key = ks + nt * 16 + l15;
                #pragma unroll
                for (int i = 0; i < 8; ++i) {
                    const int row = qrow0 + half * 8 + i;
                    if (key > row) sc[nt][i] = -1e30f;
                }
            }
        }

        // ---- online softmax: cross-lane reductions per 16-lane half ----
        float tmax[8];
        #pragma unroll
        for (int i = 0; i < 8; ++i) tmax[i] = fmaxf(sc[0][i], sc[1][i]);
        #pragma unroll
        for (int off = 1; off < 16; off <<= 1)
            #pragma unroll
            for (int i = 0; i < 8; ++i)
                tmax[i] = fmaxf(tmax[i], __shfl_xor(tmax[i], off, 16));

        float rsum[8];
        #pragma unroll
        for (int i = 0; i < 8; ++i) {
            const float mn    = fmaxf(mrow[i], tmax[i]);
            const float alpha = __expf(mrow[i] - mn);
            mrow[i] = mn;
            const float p0 = __expf(sc[0][i] - mn);
            const float p1 = __expf(sc[1][i] - mn);
            sc[0][i] = p0; sc[1][i] = p1;
            rsum[i]  = p0 + p1;
            lrow[i] *= alpha;
            #pragma unroll
            for (int t = 0; t < 4; ++t) oacc[t][i] *= alpha;
        }
        #pragma unroll
        for (int off = 1; off < 16; off <<= 1)
            #pragma unroll
            for (int i = 0; i < 8; ++i) rsum[i] += __shfl_xor(rsum[i], off, 16);
        #pragma unroll
        for (int i = 0; i < 8; ++i) lrow[i] += rsum[i];

        // ---- P: D-layout -> A-fragment via per-wave LDS round trip ----
        _Float16* pw = Ps[wave];
        #pragma unroll
        for (int nt = 0; nt < 2; ++nt) {
            const int keyl = nt * 16 + l15;
            #pragma unroll
            for (int i = 0; i < 8; ++i)
                pw[(half * 8 + i) * 32 + keyl] = (_Float16)sc[nt][i];
        }
        asm volatile("s_wait_dscnt 0x0" ::: "memory");
        v16h pf;
        {
            v8h plo = *(const v8h*)&pw[l15 * 32 + half * 8];
            v8h phi = *(const v8h*)&pw[l15 * 32 + 16 + half * 8];
            #pragma unroll
            for (int j = 0; j < 8; ++j) { pf[j] = plo[j]; pf[8 + j] = phi[j]; }
        }

        // ---- O += P @ V : V fragments contiguous from per-head-transposed Vt
        #pragma unroll
        for (int t = 0; t < 4; ++t) {
            const _Float16* vp =
                Vt + (size_t)((b * HH + h) * DHH + t * 16 + l15) * SS + ks;
            v8h vlo = *(const v8h*)(vp + half * 8);
            v8h vhi = *(const v8h*)(vp + 16 + half * 8);
            v16h vf;
            #pragma unroll
            for (int j = 0; j < 8; ++j) { vf[j] = vlo[j]; vf[8 + j] = vhi[j]; }
            oacc[t] = __builtin_amdgcn_wmma_f32_16x16x32_f16(
                false, pf, false, vf, (short)0, oacc[t], false, false);
        }
    }

    // ---- normalize + f16 store ----
    #pragma unroll
    for (int t = 0; t < 4; ++t)
        #pragma unroll
        for (int i = 0; i < 8; ++i) {
            const int row = qrow0 + half * 8 + i;
            O[(size_t)(b * Sq + row) * DD + colbase + t * 16 + l15] =
                (_Float16)(oacc[t][i] * (1.0f / lrow[i]));
        }
}

// ---------------------------------------------------------------------------
// Out = LayerNorm(X + Y)*g + b   (row = 512); also writes f16 mirror.
// ---------------------------------------------------------------------------
__global__ void __launch_bounds__(256)
add_ln(const float* __restrict__ X, const float* __restrict__ Y,
       const float* __restrict__ g, const float* __restrict__ be,
       float* __restrict__ Out, _Float16* __restrict__ OutH)
{
    __shared__ float red[16];
    const int row = blockIdx.x;
    const int tid = threadIdx.x;
    const size_t base = (size_t)row * DD;

    const float v0 = X[base + tid] + Y[base + tid];
    const float v1 = X[base + 256 + tid] + Y[base + 256 + tid];
    float s  = v0 + v1;
    float s2 = v0 * v0 + v1 * v1;
    #pragma unroll
    for (int off = 1; off < 32; off <<= 1) {
        s  += __shfl_xor(s,  off, 32);
        s2 += __shfl_xor(s2, off, 32);
    }
    const int wave = tid >> 5, lane = tid & 31;
    if (lane == 0) { red[wave] = s; red[8 + wave] = s2; }
    __syncthreads();
    if (wave == 0) {
        float a  = (lane < 8) ? red[lane] : 0.0f;
        float a2 = (lane < 8) ? red[8 + lane] : 0.0f;
        #pragma unroll
        for (int off = 1; off < 8; off <<= 1) {
            a  += __shfl_xor(a,  off, 32);
            a2 += __shfl_xor(a2, off, 32);
        }
        if (lane == 0) { red[0] = a; red[1] = a2; }
    }
    __syncthreads();
    const float mean = red[0] * (1.0f / DD);
    const float var  = red[1] * (1.0f / DD) - mean * mean;
    const float rstd = rsqrtf(var + 1e-5f);
    const float o0 = (v0 - mean) * rstd * g[tid]       + be[tid];
    const float o1 = (v1 - mean) * rstd * g[256 + tid] + be[256 + tid];
    Out[base + tid]        = o0;
    Out[base + 256 + tid]  = o1;
    OutH[base + tid]       = (_Float16)o0;
    OutH[base + 256 + tid] = (_Float16)o1;
}

__global__ void add_pos(const float* __restrict__ Xin, const float* __restrict__ P,
                        float* __restrict__ X, _Float16* __restrict__ Xh)
{
    const int idx = blockIdx.x * blockDim.x + threadIdx.x;
    const float v = Xin[idx] + P[idx % (SS * DD)];
    X[idx]  = v;
    Xh[idx] = (_Float16)v;
}

__global__ void cvt_h(const float* __restrict__ X, _Float16* __restrict__ Xh)
{
    const int idx = blockIdx.x * blockDim.x + threadIdx.x;
    Xh[idx] = (_Float16)X[idx];
}

// W[l] is [K,N] fp32 -> Wt[l] is [N,K] f16 (coalesced writes along K)
__global__ void transpose_w(const float* __restrict__ W, _Float16* __restrict__ Wt,
                            int K, int N)
{
    const size_t l = blockIdx.y;
    const int e = blockIdx.x * 256 + threadIdx.x;   // over N*K
    const int n = e / K, k = e % K;
    Wt[l * (size_t)N * K + (size_t)n * K + k] =
        (_Float16)W[l * (size_t)K * N + (size_t)k * N + n];
}

// ---------------------------------------------------------------------------
extern "C" void kernel_launch(void* const* d_in, const int* in_sizes, int n_in,
                              void* d_out, int out_size, void* d_ws, size_t ws_size,
                              hipStream_t stream)
{
    const float* kv  = (const float*)d_in[0];
    const float* xin = (const float*)d_in[1];
    const float* pos = (const float*)d_in[2];
    const float* sqW = (const float*)d_in[3];  const float* sqb = (const float*)d_in[4];
    const float* skW = (const float*)d_in[5];  const float* skb = (const float*)d_in[6];
    const float* svW = (const float*)d_in[7];  const float* svb = (const float*)d_in[8];
    const float* soW = (const float*)d_in[9];  const float* sob = (const float*)d_in[10];
    const float* cqW = (const float*)d_in[11]; const float* cqb = (const float*)d_in[12];
    const float* ckW = (const float*)d_in[13]; const float* ckb = (const float*)d_in[14];
    const float* cvW = (const float*)d_in[15]; const float* cvb = (const float*)d_in[16];
    const float* coW = (const float*)d_in[17]; const float* cob = (const float*)d_in[18];
    const float* f1W = (const float*)d_in[19]; const float* f1b = (const float*)d_in[20];
    const float* f2W = (const float*)d_in[21]; const float* f2b = (const float*)d_in[22];
    const float* g1  = (const float*)d_in[23]; const float* b1  = (const float*)d_in[24];
    const float* g2  = (const float*)d_in[25]; const float* b2  = (const float*)d_in[26];
    const float* g3  = (const float*)d_in[27]; const float* b3  = (const float*)d_in[28];
    float* out = (float*)d_out;

    // ---- workspace carve-up ----
    char* p = (char*)d_ws;
    float*    x    = (float*)p;     p += (size_t)MM * DD * 4;
    float*    y    = (float*)p;     p += (size_t)MM * DD * 4;
    _Float16* xh   = (_Float16*)p;  p += (size_t)MM * DD * 2;
    _Float16* kvh  = (_Float16*)p;  p += (size_t)MM * DD * 2;
    _Float16* qh   = (_Float16*)p;  p += (size_t)MM * DD * 2;
    _Float16* kh   = (_Float16*)p;  p += (size_t)MM * DD * 2;
    _Float16* vth  = (_Float16*)p;  p += (size_t)MM * DD * 2;
    _Float16* atth = (_Float16*)p;  p += (size_t)MM * DD * 2;
    _Float16* hh_  = (_Float16*)p;  p += (size_t)MM * FF * 2;
    _Float16* wq   = (_Float16*)p;  p += (size_t)LL * DD * DD * 2;
    _Float16* wk   = (_Float16*)p;  p += (size_t)LL * DD * DD * 2;
    _Float16* wv   = (_Float16*)p;  p += (size_t)LL * DD * DD * 2;
    _Float16* wo   = (_Float16*)p;  p += (size_t)LL * DD * DD * 2;
    _Float16* wcq  = (_Float16*)p;  p += (size_t)LL * DD * DD * 2;
    _Float16* wck  = (_Float16*)p;  p += (size_t)LL * DD * DD * 2;
    _Float16* wcv  = (_Float16*)p;  p += (size_t)LL * DD * DD * 2;
    _Float16* wco  = (_Float16*)p;  p += (size_t)LL * DD * DD * 2;
    _Float16* w1   = (_Float16*)p;  p += (size_t)LL * DD * FF * 2;
    _Float16* w2   = (_Float16*)p;  p += (size_t)LL * FF * DD * 2;

    // ---- one-time (per call) weight convert+transpose, input conversions ----
    const dim3 tD((DD * DD) / 256, LL);
    transpose_w<<<tD, 256, 0, stream>>>(sqW, wq,  DD, DD);
    transpose_w<<<tD, 256, 0, stream>>>(skW, wk,  DD, DD);
    transpose_w<<<tD, 256, 0, stream>>>(svW, wv,  DD, DD);
    transpose_w<<<tD, 256, 0, stream>>>(soW, wo,  DD, DD);
    transpose_w<<<tD, 256, 0, stream>>>(cqW, wcq, DD, DD);
    transpose_w<<<tD, 256, 0, stream>>>(ckW, wck, DD, DD);
    transpose_w<<<tD, 256, 0, stream>>>(cvW, wcv, DD, DD);
    transpose_w<<<tD, 256, 0, stream>>>(coW, wco, DD, DD);
    const dim3 tF((DD * FF) / 256, LL);
    transpose_w<<<tF, 256, 0, stream>>>(f1W, w1, DD, FF);
    transpose_w<<<tF, 256, 0, stream>>>(f2W, w2, FF, DD);

    cvt_h<<<(MM * DD) / 256, 256, 0, stream>>>(kv, kvh);
    add_pos<<<(MM * DD) / 256, 256, 0, stream>>>(xin, pos, x, xh);

    const dim3 gDD(DD / 128, MM / 128);   // 4 x 32
    const dim3 gFF(FF / 128, MM / 128);   // 16 x 32
    const int attnBlocks = BB * HH * (SS / 64);  // 512

    for (int l = 0; l < LL; ++l) {
        const size_t oD = (size_t)l * DD * DD, oF = (size_t)l * DD * FF;
        const size_t bD = (size_t)l * DD,      bF = (size_t)l * FF;

        // ---- masked self-attention ----
        gemm_wmma<<<gDD, 256, 0, stream>>>(xh, wq + oD, sqb + bD, qh,  MM, DD, DD, MODE_F16);
        gemm_wmma<<<gDD, 256, 0, stream>>>(xh, wk + oD, skb + bD, kh,  MM, DD, DD, MODE_F16);
        gemm_wmma<<<gDD, 256, 0, stream>>>(xh, wv + oD, svb + bD, vth, MM, DD, DD, MODE_VT);
        attn_wmma<<<attnBlocks, 128, 0, stream>>>(qh, kh, vth, atth, SS, SS, 1);
        gemm_wmma<<<gDD, 256, 0, stream>>>(atth, wo + oD, sob + bD, y, MM, DD, DD, MODE_F32);
        add_ln<<<MM, 256, 0, stream>>>(x, y, g1 + bD, b1 + bD, x, xh);

        // ---- cross-attention ----
        gemm_wmma<<<gDD, 256, 0, stream>>>(xh,  wcq + oD, cqb + bD, qh,  MM, DD, DD, MODE_F16);
        gemm_wmma<<<gDD, 256, 0, stream>>>(kvh, wck + oD, ckb + bD, kh,  MM, DD, DD, MODE_F16);
        gemm_wmma<<<gDD, 256, 0, stream>>>(kvh, wcv + oD, cvb + bD, vth, MM, DD, DD, MODE_VT);
        attn_wmma<<<attnBlocks, 128, 0, stream>>>(qh, kh, vth, atth, SS, SS, 0);
        gemm_wmma<<<gDD, 256, 0, stream>>>(atth, wco + oD, cob + bD, y, MM, DD, DD, MODE_F32);
        add_ln<<<MM, 256, 0, stream>>>(x, y, g2 + bD, b2 + bD, x, xh);

        // ---- FFN ----
        gemm_wmma<<<gFF, 256, 0, stream>>>(xh, w1 + oF, f1b + bF, hh_, MM, FF, DD, MODE_F16_RELU);
        gemm_wmma<<<gDD, 256, 0, stream>>>(hh_, w2 + oF, f2b + bD, y,  MM, DD, FF, MODE_F32);
        float* lnout = (l == LL - 1) ? out : x;
        add_ln<<<MM, 256, 0, stream>>>(x, y, g3 + bD, b3 + bD, lnout, xh);
    }
}
